// LoRALinear_40355512713642
// MI455X (gfx1250) — matrix operand order repaired and checked by reading, not compiled, not run on hardware
//
#include <hip/hip_runtime.h>
#include <hip/hip_bf16.h>
#include <cstdint>
#include <cstddef>

// ---------------------------------------------------------------------------
// LoRA linear: out = x @ W^T + 3 * (x @ B^T) @ A^T + bias
//   x      [8192, 4096] f32      weight [4096, 4096] f32
//   bias   [4096] f32            lora_a [4096, 16] f32   lora_b [16, 4096] f32
//
// Fast path (needs ~203 MB d_ws):
//   k1: low3 = 3 * x @ B^T                                  (8192 x 16, f32)
//   k2: split x  -> bf16 hi/lo planes, width 4128 (cols 4096..4111 = low3,
//       4112..4127 = 0)   k3: same for W with lora_a appended
//   k4: bf16x3 GEMM (v_wmma_f32_16x16x32_bf16), K = 4128 uniform, LDS
//       double-buffered, staging via GLOBAL_LOAD_ASYNC_TO_LDS_B128 (ASYNCcnt),
//       one s_wait_asynccnt + barrier per K-step. Nontemporal output stores.
// Fallback path (small d_ws): in-loop fp32->bf16 split GEMM.
// ---------------------------------------------------------------------------

typedef __attribute__((ext_vector_type(16))) __bf16 v16bf;
typedef __attribute__((ext_vector_type(8)))  float  v8f;

#define T_DIM 8192
#define K_DIM 4096
#define N_DIM 4096
#define RANK  16
#define AUGK  4128            // K_DIM + 32 (16 LoRA cols + 16 zero cols)
#define LDSK  40              // padded LDS row stride in ushorts (80B, 16B-aligned)
#define TILE_US (128 * LDSK)  // one 128x32 bf16 tile (padded), in ushorts

// Types for the async-to-LDS builtin: pointee is an int4 vector; global side
// lives in AS1, LDS side in AS3 (C-style cast -> addrspacecast).
typedef int v4i __attribute__((vector_size(16)));
typedef __attribute__((address_space(1))) v4i as1_v4i;
typedef __attribute__((address_space(3))) v4i as3_v4i;

// ---------------- bf16 split helpers (hi = trunc, lo = RNE(residual)) -------
union Pack16 { unsigned int u[8]; uint4 q[2]; };

static __device__ __forceinline__ void split_pack16(const float* v,
                                                    Pack16& H, Pack16& L) {
#pragma unroll
  for (int j = 0; j < 8; ++j) {
    float f0 = v[2 * j + 0];
    float f1 = v[2 * j + 1];
    unsigned int u0 = __float_as_uint(f0);
    unsigned int u1 = __float_as_uint(f1);
    unsigned int h0 = u0 & 0xFFFF0000u;
    unsigned int h1 = u1 & 0xFFFF0000u;
    H.u[j] = (h0 >> 16) | h1;
    float r0 = f0 - __uint_as_float(h0);   // exact residual
    float r1 = f1 - __uint_as_float(h1);
    unsigned int t0 = __float_as_uint(r0);
    unsigned int t1 = __float_as_uint(r1);
    t0 += 0x7fffu + ((t0 >> 16) & 1u);     // RNE
    t1 += 0x7fffu + ((t1 >> 16) & 1u);
    L.u[j] = (t0 >> 16) | (t1 & 0xFFFF0000u);
  }
}

static __device__ __forceinline__ void split_store16_lds(unsigned short* hp,
                                                         unsigned short* lp,
                                                         const float* v) {
  Pack16 H, L;
  split_pack16(v, H, L);
  *(uint4*)(hp)     = H.q[0];
  *(uint4*)(hp + 8) = H.q[1];
  *(uint4*)(lp)     = L.q[0];
  *(uint4*)(lp + 8) = L.q[1];
}

// ---------------- WMMA fragment helpers -------------------------------------
union FragU { v16bf v; uint4 q[2]; };

// A fragment (16x32 bf16): lane L -> row (L&15); K chunks at (L>>4)*8, 16+(L>>4)*8
static __device__ __forceinline__ v16bf load_afrag(const unsigned short* base,
                                                   int rowBase, int lane) {
  const unsigned short* p = base + (rowBase + (lane & 15)) * LDSK + ((lane >> 4) << 3);
  FragU f;
  f.q[0] = *(const uint4*)(p);
  f.q[1] = *(const uint4*)(p + 16);
  return f.v;
}

// B fragment (32x16 bf16): lane L -> col (L&15); K chunks at (L>>4)*8, 16+(L>>4)*8
static __device__ __forceinline__ v16bf load_bfrag(const unsigned short* base,
                                                   int colBase, int lane) {
  const unsigned short* p = base + (colBase + (lane & 15)) * LDSK + ((lane >> 4) << 3);
  FragU f;
  f.q[0] = *(const uint4*)(p);
  f.q[1] = *(const uint4*)(p + 16);
  return f.v;
}

// bf16x3: lo*hi + hi*lo + hi*hi into the f32 accumulator
static __device__ __forceinline__ void wmma3(v8f& c, v16bf ah, v16bf al,
                                             v16bf bh, v16bf bl) {
  c = __builtin_amdgcn_wmma_f32_16x16x32_bf16(false, al, false, bh, (short)0, c, false, false);
  c = __builtin_amdgcn_wmma_f32_16x16x32_bf16(false, ah, false, bl, (short)0, c, false, false);
  c = __builtin_amdgcn_wmma_f32_16x16x32_bf16(false, ah, false, bh, (short)0, c, false, false);
}

// shared compute: 4x2 wmma tiles per wave from one 4-tile LDS buffer
static __device__ __forceinline__ void compute_tiles(const unsigned short* b,
                                                     int mw, int nw, int lane,
                                                     v8f (&acc)[4][2]) {
  v16bf ah[4], al[4], bh[2], bl[2];
#pragma unroll
  for (int i = 0; i < 4; ++i) {
    ah[i] = load_afrag(b + 0 * TILE_US, mw * 64 + i * 16, lane);
    al[i] = load_afrag(b + 1 * TILE_US, mw * 64 + i * 16, lane);
  }
#pragma unroll
  for (int j = 0; j < 2; ++j) {
    bh[j] = load_bfrag(b + 2 * TILE_US, nw * 32 + j * 16, lane);
    bl[j] = load_bfrag(b + 3 * TILE_US, nw * 32 + j * 16, lane);
  }
#pragma unroll
  for (int i = 0; i < 4; ++i)
#pragma unroll
    for (int j = 0; j < 2; ++j)
      wmma3(acc[i][j], ah[i], al[i], bh[j], bl[j]);
}

// shared epilogue: C/D layout col = lane&15, row = v + (lane>>4)*8; NT stores
static __device__ __forceinline__ void epilogue_store(v8f (&acc)[4][2],
                                                      const float* bias,
                                                      float* out, int m0, int n0,
                                                      int mw, int nw, int lane) {
#pragma unroll
  for (int j = 0; j < 2; ++j) {
    const int col = n0 + nw * 32 + j * 16 + (lane & 15);
    const float bv = bias[col];
#pragma unroll
    for (int i = 0; i < 4; ++i) {
      const int rowBase = m0 + mw * 64 + i * 16 + ((lane >> 4) << 3);
#pragma unroll
      for (int v = 0; v < 8; ++v)
        __builtin_nontemporal_store(acc[i][j][v] + bv,
                                    &out[(size_t)(rowBase + v) * N_DIM + col]);
    }
  }
}

// async 16B global->LDS copy (gfx1250, tracked on ASYNCcnt)
static __device__ __forceinline__ void async16(const unsigned short* g,
                                               unsigned short* l) {
  __builtin_amdgcn_global_load_async_to_lds_b128(
      (as1_v4i*)g, (as3_v4i*)l, 0, 0);
}

// ---------------------------------------------------------------------------
// Kernel 1: low3[t][r] = 3 * sum_k x[t][k] * lora_b[r][k]
// ---------------------------------------------------------------------------
__global__ __launch_bounds__(256) void lora_low_kernel(
    const float* __restrict__ x, const float* __restrict__ lora_b,
    float* __restrict__ low3) {
  const int lane = threadIdx.x & 31;
  const int t = blockIdx.x * 8 + (threadIdx.x >> 5);

  float acc[RANK];
#pragma unroll
  for (int r = 0; r < RANK; ++r) acc[r] = 0.0f;

  const float* xrow = x + (size_t)t * K_DIM;
  for (int kb = lane * 4; kb < K_DIM; kb += 128) {
    float4 xv = *(const float4*)(xrow + kb);
#pragma unroll
    for (int r = 0; r < RANK; ++r) {
      float4 bv = *(const float4*)(lora_b + (size_t)r * K_DIM + kb);
      acc[r] += xv.x * bv.x + xv.y * bv.y + xv.z * bv.z + xv.w * bv.w;
    }
  }
#pragma unroll
  for (int off = 16; off > 0; off >>= 1) {
#pragma unroll
    for (int r = 0; r < RANK; ++r)
      acc[r] += __shfl_xor(acc[r], off, 32);
  }
  if (lane < RANK) low3[(size_t)t * RANK + lane] = 3.0f * acc[lane];
}

// ---------------------------------------------------------------------------
// Kernels 2/3: split fp32 matrix (+16 aug cols, +16 zero cols) into bf16
// hi/lo planes of width AUGK. One thread per 16-column chunk.
// ---------------------------------------------------------------------------
__global__ __launch_bounds__(256) void split_convert_kernel(
    const float* __restrict__ src, const float* __restrict__ aug,
    unsigned short* __restrict__ hp, unsigned short* __restrict__ lp,
    int rows, int srcStride, int augStride) {
  const int CH = AUGK / 16;                       // 258 chunks per row
  int idx = blockIdx.x * 256 + threadIdx.x;
  if (idx >= rows * CH) return;
  int row = idx / CH;
  int cc  = idx - row * CH;
  int c0  = cc * 16;

  float v[16];
  if (cc < CH - 2) {                              // main matrix columns
    const float* sp = src + (size_t)row * srcStride + c0;
#pragma unroll
    for (int q = 0; q < 4; ++q) {
      float4 a = *(const float4*)(sp + 4 * q);
      v[4 * q + 0] = a.x; v[4 * q + 1] = a.y; v[4 * q + 2] = a.z; v[4 * q + 3] = a.w;
    }
  } else if (cc == CH - 2) {                      // LoRA augmentation columns
    const float* sp = aug + (size_t)row * augStride;
#pragma unroll
    for (int q = 0; q < 4; ++q) {
      float4 a = *(const float4*)(sp + 4 * q);
      v[4 * q + 0] = a.x; v[4 * q + 1] = a.y; v[4 * q + 2] = a.z; v[4 * q + 3] = a.w;
    }
  } else {                                        // zero padding columns
#pragma unroll
    for (int j = 0; j < 16; ++j) v[j] = 0.0f;
  }

  Pack16 H, L;
  split_pack16(v, H, L);
  uint4* hq = (uint4*)(hp + (size_t)row * AUGK + c0);
  uint4* lq = (uint4*)(lp + (size_t)row * AUGK + c0);
  hq[0] = H.q[0]; hq[1] = H.q[1];
  lq[0] = L.q[0]; lq[1] = L.q[1];
}

// ---------------------------------------------------------------------------
// Kernel 4 (fast path): bf16x3 GEMM from pre-split planes. 128x128 tile per
// 256-thread block (8 waves); K = AUGK, uniform. LDS double-buffered; tiles
// staged with GLOBAL_LOAD_ASYNC_TO_LDS_B128; one asynccnt-wait + barrier per
// K-step. No conversion VALU in the loop at all.
// ---------------------------------------------------------------------------
__global__ __launch_bounds__(256) void lora_gemm_pre_kernel(
    const unsigned short* __restrict__ xh, const unsigned short* __restrict__ xl,
    const unsigned short* __restrict__ wh, const unsigned short* __restrict__ wl,
    const float* __restrict__ bias, float* __restrict__ out) {
  extern __shared__ uint4 smem4[];                // 2 bufs * 4 tiles = 80 KB
  unsigned short* smem = (unsigned short*)smem4;

  const int tid  = threadIdx.x;
  const int lane = tid & 31;
  const int wave = tid >> 5;
  const int mw = wave >> 2, nw = wave & 3;
  const int m0 = blockIdx.y * 128, n0 = blockIdx.x * 128;
  const int srow = tid >> 1;                      // staging row 0..127
  const int kh   = (tid & 1) << 4;                // staging K half: 0 or 16

  v8f acc[4][2];
#pragma unroll
  for (int i = 0; i < 4; ++i)
#pragma unroll
    for (int j = 0; j < 2; ++j)
      acc[i][j] = (v8f){0.f, 0.f, 0.f, 0.f, 0.f, 0.f, 0.f, 0.f};

  const int KT2 = AUGK / 32;                      // 129 uniform K-steps

  auto stage_async = [&](int p, int kt) {
    unsigned short* b = smem + p * 4 * TILE_US;
    const size_t gx = (size_t)(m0 + srow) * AUGK + kt * 32 + kh;
    const size_t gw = (size_t)(n0 + srow) * AUGK + kt * 32 + kh;
    unsigned short* d0 = b + 0 * TILE_US + srow * LDSK + kh;
    unsigned short* d1 = b + 1 * TILE_US + srow * LDSK + kh;
    unsigned short* d2 = b + 2 * TILE_US + srow * LDSK + kh;
    unsigned short* d3 = b + 3 * TILE_US + srow * LDSK + kh;
    async16(xh + gx, d0); async16(xh + gx + 8, d0 + 8);
    async16(xl + gx, d1); async16(xl + gx + 8, d1 + 8);
    async16(wh + gw, d2); async16(wh + gw + 8, d2 + 8);
    async16(wl + gw, d3); async16(wl + gw + 8, d3 + 8);
  };

  stage_async(0, 0);
  asm volatile("s_wait_asynccnt 0x0" ::: "memory");
  __syncthreads();

  for (int kt = 0; kt < KT2; ++kt) {
    const int p = kt & 1;
    if (kt + 1 < KT2) stage_async(p ^ 1, kt + 1);   // DMA next tile while computing
    compute_tiles(smem + p * 4 * TILE_US, mw, nw, lane, acc);
    asm volatile("s_wait_asynccnt 0x0" ::: "memory");
    __syncthreads();
  }

  epilogue_store(acc, bias, out, m0, n0, mw, nw, lane);
}

// ---------------------------------------------------------------------------
// Fallback GEMM (small d_ws): in-loop fp32->bf16 split.
// ---------------------------------------------------------------------------
__global__ __launch_bounds__(256) void lora_gemm_kernel(
    const float* __restrict__ x, const float* __restrict__ weight,
    const float* __restrict__ bias, const float* __restrict__ lora_a,
    const float* __restrict__ low3, float* __restrict__ out) {
  extern __shared__ uint4 smem4[];
  unsigned short* smem = (unsigned short*)smem4;

  const int tid  = threadIdx.x;
  const int lane = tid & 31;
  const int wave = tid >> 5;
  const int mw = wave >> 2, nw = wave & 3;
  const int m0 = blockIdx.y * 128, n0 = blockIdx.x * 128;
  const int srow = tid >> 1;
  const int kh   = (tid & 1) << 4;

  v8f acc[4][2];
#pragma unroll
  for (int i = 0; i < 4; ++i)
#pragma unroll
    for (int j = 0; j < 2; ++j)
      acc[i][j] = (v8f){0.f, 0.f, 0.f, 0.f, 0.f, 0.f, 0.f, 0.f};

  float xv[16], wv[16];
  const int KT = K_DIM / 32;

  auto load_stage = [&](int kt) {
    if (kt < KT) {
      const float* xp = x      + (size_t)(m0 + srow) * K_DIM + kt * 32 + kh;
      const float* wp = weight + (size_t)(n0 + srow) * K_DIM + kt * 32 + kh;
#pragma unroll
      for (int q = 0; q < 4; ++q) {
        float4 a = *(const float4*)(xp + 4 * q);
        float4 b = *(const float4*)(wp + 4 * q);
        xv[4 * q + 0] = a.x; xv[4 * q + 1] = a.y; xv[4 * q + 2] = a.z; xv[4 * q + 3] = a.w;
        wv[4 * q + 0] = b.x; wv[4 * q + 1] = b.y; wv[4 * q + 2] = b.z; wv[4 * q + 3] = b.w;
      }
    } else if (kh == 0) {
      const float* lp = low3   + (size_t)(m0 + srow) * RANK;
      const float* ap = lora_a + (size_t)(n0 + srow) * RANK;
#pragma unroll
      for (int q = 0; q < 4; ++q) {
        float4 a = ((const float4*)lp)[q];
        float4 b = ((const float4*)ap)[q];
        xv[4 * q + 0] = a.x; xv[4 * q + 1] = a.y; xv[4 * q + 2] = a.z; xv[4 * q + 3] = a.w;
        wv[4 * q + 0] = b.x; wv[4 * q + 1] = b.y; wv[4 * q + 2] = b.z; wv[4 * q + 3] = b.w;
      }
    } else {
#pragma unroll
      for (int j = 0; j < 16; ++j) { xv[j] = 0.0f; wv[j] = 0.0f; }
    }
  };

  auto stage_store = [&](int p) {
    unsigned short* b = smem + p * 4 * TILE_US;
    split_store16_lds(b + 0 * TILE_US + srow * LDSK + kh,
                      b + 1 * TILE_US + srow * LDSK + kh, xv);
    split_store16_lds(b + 2 * TILE_US + srow * LDSK + kh,
                      b + 3 * TILE_US + srow * LDSK + kh, wv);
  };

  load_stage(0);
  stage_store(0);
  __syncthreads();
  load_stage(1);

  for (int kt = 0; kt <= KT; ++kt) {
    const int p = kt & 1;
    compute_tiles(smem + p * 4 * TILE_US, mw, nw, lane, acc);
    if (kt < KT) {
      stage_store(p ^ 1);
      if (kt + 2 <= KT) load_stage(kt + 2);
      __syncthreads();
    }
  }

  epilogue_store(acc, bias, out, m0, n0, mw, nw, lane);
}

// ---------------------------------------------------------------------------
extern "C" void kernel_launch(void* const* d_in, const int* in_sizes, int n_in,
                              void* d_out, int out_size, void* d_ws, size_t ws_size,
                              hipStream_t stream) {
  const float* x      = (const float*)d_in[0];
  const float* weight = (const float*)d_in[1];
  const float* bias   = (const float*)d_in[2];
  const float* lora_a = (const float*)d_in[3];
  const float* lora_b = (const float*)d_in[4];
  float* out = (float*)d_out;

  // d_ws layout: [low3 f32][xh][xl][wh][wl] (bf16 planes, width AUGK)
  const size_t low3Bytes = (size_t)T_DIM * RANK * sizeof(float);       // 512 KB
  const size_t xPlane    = (size_t)T_DIM * AUGK * sizeof(unsigned short);
  const size_t wPlane    = (size_t)N_DIM * AUGK * sizeof(unsigned short);
  const size_t need      = low3Bytes + 2 * xPlane + 2 * wPlane;        // ~203 MB

  float* low3 = (float*)d_ws;
  const size_t shmem = 2u * 4u * TILE_US * sizeof(unsigned short);     // 80 KB

  lora_low_kernel<<<T_DIM / 8, 256, 0, stream>>>(x, lora_b, low3);

  if (ws_size >= need) {
    unsigned short* xh = (unsigned short*)((char*)d_ws + low3Bytes);
    unsigned short* xl = (unsigned short*)((char*)xh + xPlane);
    unsigned short* wh = (unsigned short*)((char*)xl + xPlane);
    unsigned short* wl = (unsigned short*)((char*)wh + wPlane);

    const int CH = AUGK / 16;  // 258
    split_convert_kernel<<<(T_DIM * CH + 255) / 256, 256, 0, stream>>>(
        x, low3, xh, xl, T_DIM, K_DIM, RANK);
    split_convert_kernel<<<(N_DIM * CH + 255) / 256, 256, 0, stream>>>(
        weight, lora_a, wh, wl, N_DIM, K_DIM, RANK);

    // grid.x = N tiles (fastest): consecutive blocks share an x slab and sweep
    // the bf16 W planes (67 MB) which stay resident in the 192 MB L2.
    lora_gemm_pre_kernel<<<dim3(N_DIM / 128, T_DIM / 128), 256, shmem, stream>>>(
        xh, xl, wh, wl, bias, out);
  } else {
    lora_gemm_kernel<<<dim3(N_DIM / 128, T_DIM / 128), 256, shmem, stream>>>(
        x, weight, bias, lora_a, low3, out);
  }
}
